// NeuralOptimalTransport_56573309223313
// MI455X (gfx1250) — matrix operand order repaired and checked by reading, not compiled, not run on hardware
//
#include <hip/hip_runtime.h>
#include <hip/hip_bf16.h>

// Problem constants (match reference)
#define NN_     1024
#define EMBED_  256
#define HID_    128
#define EPS_    1e-8f

typedef __attribute__((ext_vector_type(16))) _Float16 v16h;
typedef __attribute__((ext_vector_type(8)))  _Float16 h8;
typedef __attribute__((ext_vector_type(4)))  _Float16 h4;
typedef __attribute__((ext_vector_type(8)))  float    v8f;

// ---------------------------------------------------------------------------
// Kernel 0: one-shot precision/layout prep.
//   blocks 0..255   : srcH[m][k]  = f16(source)           (row-major, contiguous k)
//   blocks 256..511 : tgtH[m][k]  = f16(target)
//   blocks 512..575 : W1tH[mat][n][k] = f16(W1[mat*E + k][n])   (transposed)
// Gives the WMMA kernel contiguous, layout-native fragment loads.
// ---------------------------------------------------------------------------
__global__ __launch_bounds__(256) void k_prep(
    const float* __restrict__ src, const float* __restrict__ tgt,
    const float* __restrict__ W1,
    _Float16* __restrict__ srcH, _Float16* __restrict__ tgtH,
    _Float16* __restrict__ W1tH) {
  const int b   = blockIdx.x;
  const int tid = threadIdx.x;
  if (b < 512) {
    const float* __restrict__ in = (b < 256) ? src : tgt;
    _Float16* __restrict__ op    = (b < 256) ? srcH : tgtH;
    const int base = (b & 255) * 1024 + tid * 4;   // 256 blocks * 1024 = 262144 elems
    const float4 f = *(const float4*)(in + base);
    h4 o;
    o[0] = (_Float16)f.x; o[1] = (_Float16)f.y;
    o[2] = (_Float16)f.z; o[3] = (_Float16)f.w;
    *(h4*)(op + base) = o;
  } else {
    const int e = (b - 512) * 1024 + tid * 4;      // 0..65535, 4 consecutive k
    const int m = e >> 15;                         // mat 0/1
    const int r = e & 32767;
    const int n = r >> 8;                          // 0..127
    const int k = r & 255;                         // multiple of 4
    h4 o;
#pragma unroll
    for (int q = 0; q < 4; ++q)
      o[q] = (_Float16)W1[(m * EMBED_ + k + q) * HID_ + n];
    *(h4*)(W1tH + e) = o;
  }
}

// ---------------------------------------------------------------------------
// Kernel 1: hs = source @ W1[:E], ht = target @ W1[E:], via v_wmma_f32_16x16x32_f16.
// One wave per 16x16 output tile. 2 matrices * 64 m-tiles * 8 n-tiles = 1024 tiles.
// A layout (16x32 f16): lane = M (lane&15), element e -> K = (e<8 ? e : e+8) + (lane>>4)*8
//   => two contiguous 8-half runs per lane: [hi*8, hi*8+7] and [16+hi*8, 23+hi*8]
// B layout (32x16 f16): lane = N (lane&15), element e -> K = (e<8 ? e : e+8) + (lane>>4)*8
//   => two contiguous 8-half runs of the transposed weight row (same as A)
// C/D  (16x16 f32):     lane = N, VGPR r -> M = r + (lane>>4)*8
// ---------------------------------------------------------------------------
__global__ __launch_bounds__(128) void k_gemm_hs_ht(
    const _Float16* __restrict__ srcH, const _Float16* __restrict__ tgtH,
    const _Float16* __restrict__ W1tH,
    float* __restrict__ hs, float* __restrict__ ht) {
  const int wave = threadIdx.x >> 5;
  const int lane = threadIdx.x & 31;
  const int t    = blockIdx.x * 4 + wave;     // 0..1023
  const int mat  = t >> 9;                    // 0 = source, 1 = target
  const int rem  = t & 511;
  const int m0   = (rem >> 3) << 4;           // 0..1008
  const int n0   = (rem & 7) << 4;            // 0..112

  const int mr = lane & 15;
  const int hi = lane >> 4;

  const _Float16* __restrict__ Ar =
      (mat ? tgtH : srcH) + (m0 + mr) * EMBED_;
  const _Float16* __restrict__ Wr =
      W1tH + mat * (HID_ * EMBED_) + (n0 + mr) * EMBED_;
  float* __restrict__ O = mat ? ht : hs;

  v8f acc = {};
#pragma unroll
  for (int k0 = 0; k0 < EMBED_; k0 += 32) {
    const h8 alo = *(const h8*)(Ar + k0 + hi * 8);        // e = 0..7
    const h8 ahi = *(const h8*)(Ar + k0 + 16 + hi * 8);   // e = 8..15
    const v16h a = __builtin_shufflevector(alo, ahi,
        0, 1, 2, 3, 4, 5, 6, 7, 8, 9, 10, 11, 12, 13, 14, 15);
    const h8 blo = *(const h8*)(Wr + k0 + hi * 8);
    const h8 bhi = *(const h8*)(Wr + k0 + 16 + hi * 8);
    const v16h bv = __builtin_shufflevector(blo, bhi,
        0, 1, 2, 3, 4, 5, 6, 7, 8, 9, 10, 11, 12, 13, 14, 15);
    acc = __builtin_amdgcn_wmma_f32_16x16x32_f16(
        /*neg_a=*/false, a, /*neg_b=*/false, bv,
        /*c_mod=*/(short)0, acc, /*reuse_a=*/false, /*reuse_b=*/false);
  }
#pragma unroll
  for (int r = 0; r < 8; ++r) {
    O[(m0 + r + hi * 8) * HID_ + n0 + mr] = acc[r];
  }
}

// ---------------------------------------------------------------------------
// Kernel 2: fused cost:  C[i,j] = relu(hs[i]+ht[j]+b1) . W2 + b2
//           K = exp(-C), also stores K^T for coalesced column matvec.
// 64x64 (i,j) tile per 256-thread block; hs / (ht+b1) staged in LDS as float4.
// Each thread owns a 4x4 microtile; inner loop over HID in float4 steps.
// ---------------------------------------------------------------------------
__global__ __launch_bounds__(256) void k_cost(
    const float* __restrict__ hs, const float* __restrict__ ht,
    const float* __restrict__ b1, const float* __restrict__ W2,
    const float* __restrict__ b2, float* __restrict__ Cout,
    float* __restrict__ K, float* __restrict__ KT) {
  __shared__ float4 sA[64][32];   // hs tile          (32 KB)
  __shared__ float4 sB[64][32];   // ht tile + b1     (32 KB)
  __shared__ float4 sw[32];       // W2

  const int i0  = (blockIdx.x >> 4) << 6;
  const int j0  = (blockIdx.x & 15) << 6;
  const int tid = threadIdx.x;

  const float4* hs4 = (const float4*)hs;
  const float4* ht4 = (const float4*)ht;
  const float4* b14 = (const float4*)b1;
#pragma unroll
  for (int it = 0; it < 8; ++it) {
    const int idx = tid + it * 256;         // 0..2047
    const int row = idx >> 5, c = idx & 31; // HID_/4 == 32 float4 per row
    sA[row][c] = hs4[(i0 + row) * 32 + c];
    float4 tt = ht4[(j0 + row) * 32 + c];
    const float4 bb = b14[c];
    tt.x += bb.x; tt.y += bb.y; tt.z += bb.z; tt.w += bb.w;
    sB[row][c] = tt;
  }
  if (tid < 32) sw[tid] = ((const float4*)W2)[tid];
  __syncthreads();

  const int il = (tid >> 4) << 2;   // local i base (0..60)
  const int jl = (tid & 15) << 2;   // local j base (0..60)

  float acc[4][4] = {};
  for (int h = 0; h < 32; ++h) {
    const float4 w = sw[h];
    float4 Ar[4], Bc[4];
#pragma unroll
    for (int r = 0; r < 4; ++r) Ar[r] = sA[il + r][h];
#pragma unroll
    for (int c = 0; c < 4; ++c) Bc[c] = sB[jl + c][h];
#pragma unroll
    for (int r = 0; r < 4; ++r)
#pragma unroll
      for (int c = 0; c < 4; ++c) {
        acc[r][c] += fmaxf(Ar[r].x + Bc[c].x, 0.0f) * w.x
                   + fmaxf(Ar[r].y + Bc[c].y, 0.0f) * w.y
                   + fmaxf(Ar[r].z + Bc[c].z, 0.0f) * w.z
                   + fmaxf(Ar[r].w + Bc[c].w, 0.0f) * w.w;
      }
  }

  const float bb2 = b2[0];
#pragma unroll
  for (int r = 0; r < 4; ++r) {
    const int gi = i0 + il + r;
#pragma unroll
    for (int c = 0; c < 4; ++c) {
      const int gj = j0 + jl + c;
      const float Cv = acc[r][c] + bb2;
      const float Kv = expf(-Cv);
      Cout[gi * NN_ + gj] = Cv;
      K[gi * NN_ + gj]    = Kv;
      KT[gj * NN_ + gi]   = Kv;
    }
  }
}

// ---------------------------------------------------------------------------
// Sinkhorn: the reference's fori_loop body ignores its carry (uses the outer
// constant a,b every iteration), so 100 iterations == 1 iteration:
//   u = (1/n) / ((1/n) * rowsum(K) + eps)
//   v = (1/n) / (K^T @ u + eps)
// Kernels 3/4: 16 lanes per row, float4 loads, shfl_xor width-16 reduction.
// ---------------------------------------------------------------------------
__global__ __launch_bounds__(256) void k_u(const float* __restrict__ K,
                                           float* __restrict__ u) {
  const int g   = blockIdx.x * 256 + threadIdx.x;  // 0..16383
  const int row = g >> 4, sub = g & 15;
  const float4* Kr = (const float4*)(K + row * NN_);
  float s = 0.0f;
#pragma unroll
  for (int q = 0; q < 16; ++q) {
    const float4 k4 = Kr[sub * 16 + q];
    s += k4.x + k4.y + k4.z + k4.w;
  }
  s += __shfl_xor(s, 8, 16);
  s += __shfl_xor(s, 4, 16);
  s += __shfl_xor(s, 2, 16);
  s += __shfl_xor(s, 1, 16);
  if (sub == 0) {
    const float inv_n = 1.0f / (float)NN_;
    u[row] = inv_n / (inv_n * s + EPS_);
  }
}

__global__ __launch_bounds__(256) void k_v(const float* __restrict__ KT,
                                           const float* __restrict__ u,
                                           float* __restrict__ v) {
  const int g   = blockIdx.x * 256 + threadIdx.x;
  const int row = g >> 4, sub = g & 15;
  const float4* Kr = (const float4*)(KT + row * NN_);
  const float4* u4 = (const float4*)u;
  float s = 0.0f;
#pragma unroll
  for (int q = 0; q < 16; ++q) {
    const float4 k4 = Kr[sub * 16 + q];
    const float4 uu = u4[sub * 16 + q];
    s += k4.x * uu.x + k4.y * uu.y + k4.z * uu.z + k4.w * uu.w;
  }
  s += __shfl_xor(s, 8, 16);
  s += __shfl_xor(s, 4, 16);
  s += __shfl_xor(s, 2, 16);
  s += __shfl_xor(s, 1, 16);
  if (sub == 0) {
    const float inv_n = 1.0f / (float)NN_;
    v[row] = inv_n / (s + EPS_);
  }
}

// ---------------------------------------------------------------------------
// Kernel 5: plan[i,j] = u[i] * K[i,j] * v[j], float4 streaming.
// ---------------------------------------------------------------------------
__global__ __launch_bounds__(256) void k_plan(const float* __restrict__ K,
                                              const float* __restrict__ u,
                                              const float* __restrict__ v,
                                              float* __restrict__ out) {
  const int g  = blockIdx.x * 256 + threadIdx.x;  // 0..262143 float4 slots
  const int i  = g >> 8;                          // 256 float4 per row
  const int c4 = g & 255;
  const float4 k4 = ((const float4*)K)[g];
  const float4 v4 = ((const float4*)v)[c4];
  const float ui  = u[i];
  float4 o;
  o.x = ui * k4.x * v4.x;
  o.y = ui * k4.y * v4.y;
  o.z = ui * k4.z * v4.z;
  o.w = ui * k4.w * v4.w;
  ((float4*)out)[g] = o;
}

// ---------------------------------------------------------------------------
// Launch. Inputs: source, target, W1, b1, W2, b2 (all f32).
// d_out = [transport_plan (1M f32)][C (1M f32)].
// Workspace layout (float-slot offsets), fully rewritten every launch:
//   hs   @ 0        (131072)      ht   @ 131072  (131072)
//   K    @ 262144   (1048576)     KT   @ 1310720 (1048576)
//   u    @ 2359296  (1024)        v    @ 2360320 (1024)
//   srcH @ 2361344  (131072 slots = 262144 f16)
//   tgtH @ 2492416  (131072 slots)
//   W1tH @ 2623488  (32768 slots = 65536 f16)
//   total 2656256 floats ~= 10.6 MB
// ---------------------------------------------------------------------------
extern "C" void kernel_launch(void* const* d_in, const int* in_sizes, int n_in,
                              void* d_out, int out_size, void* d_ws, size_t ws_size,
                              hipStream_t stream) {
  const float* src = (const float*)d_in[0];
  const float* tgt = (const float*)d_in[1];
  const float* W1  = (const float*)d_in[2];
  const float* b1  = (const float*)d_in[3];
  const float* W2  = (const float*)d_in[4];
  const float* b2  = (const float*)d_in[5];
  float* out = (float*)d_out;

  float* ws = (float*)d_ws;
  float*     hs   = ws;
  float*     ht   = ws + 131072;
  float*     K    = ws + 262144;
  float*     KT   = ws + 1310720;
  float*     u    = ws + 2359296;
  float*     v    = ws + 2360320;
  _Float16*  srcH = (_Float16*)(ws + 2361344);
  _Float16*  tgtH = (_Float16*)(ws + 2492416);
  _Float16*  W1tH = (_Float16*)(ws + 2623488);

  float* Cout = out + (size_t)NN_ * NN_;   // second output
  float* plan = out;                       // first output

  k_prep<<<576, 256, 0, stream>>>(src, tgt, W1, srcH, tgtH, W1tH);
  k_gemm_hs_ht<<<256, 128, 0, stream>>>(srcH, tgtH, W1tH, hs, ht);
  k_cost<<<256, 256, 0, stream>>>(hs, ht, b1, W2, b2, Cout, K, KT);
  k_u<<<64, 256, 0, stream>>>(K, u);
  k_v<<<64, 256, 0, stream>>>(KT, u, v);
  k_plan<<<1024, 256, 0, stream>>>(K, u, v, plan);
}
